// FNO_71648644432355
// MI455X (gfx1250) — matrix operand order repaired and checked by reading, not compiled, not run on hardware
//
#include <hip/hip_runtime.h>
#include <hip/hip_bf16.h>
#include <math.h>

// ---------------------------------------------------------------------------
// FNO 2D spectral conv as truncated-DFT GEMM chain on CDNA5 WMMA (gfx1250).
// B=32, Cin=Cout=32, H=W=128, M1=M2=16. Only 32x16 of 128x65 rfft2 modes kept
// -> every stage is a small GEMM on v_wmma_f32_16x16x32_f16 (f16 in, f32 acc).
// Roofline: 64MB in + 64MB out ~ 5.5us @ 23.3 TB/s; ~3.4 GFLOP of GEMM work,
// so WMMA pushes compute far below the bandwidth floor.
// All WMMA operands are laid out so each lane's fragment is contiguous in LDS:
//   A  row-major  [m][k]  -> one 32B v16h load (ds_load_b128 x2)
//   B  TRANSPOSED [n][k]  -> one 32B v16h load (ds_load_b128 x2)
//   C/D stores vectorized (v8h to LDS, v8f to global for the final stage).
// ---------------------------------------------------------------------------

typedef __attribute__((ext_vector_type(16))) _Float16 v16h;
typedef __attribute__((ext_vector_type(8)))  _Float16 v8h;
typedef __attribute__((ext_vector_type(4)))  _Float16 v4h;
typedef __attribute__((ext_vector_type(8)))  float    v8f;

#define NFREQ 512   // 32 kept kx * 16 ky
#define PLANE 1024  // 32*32 (b*i, b*o, or o*i) per frequency

// A fragment (16x32 f16, MxK, row-major [m][k]): lane row m=lane&15;
// lanes 0-15: K=0..15; lanes 16-31: K=16..31.
__device__ __forceinline__ v16h ldsA(const _Float16* p, int r0, int c0, int ld, int lane) {
  const int g = lane >> 4, m = lane & 15;
  const _Float16* row = p + (r0 + m) * ld + c0 + 16 * g;
  return *(const v16h*)(row);
}
// B fragment (32x16 f16, KxN) from TRANSPOSED storage Bt[n][k]: lane col
// n=lane&15; lanes 0-15 hold K=0..15, lanes 16-31 hold K=16..31 -> one 32B run.
__device__ __forceinline__ v16h ldsBt(const _Float16* p, int n0, int k0, int ldk, int lane) {
  const int g = lane >> 4, n = lane & 15;
  return *(const v16h*)(p + (n0 + n) * ldk + k0 + 16 * g);
}
__device__ __forceinline__ v8f wmma16(v16h a, v16h b, v8f c) {
  return __builtin_amdgcn_wmma_f32_16x16x32_f16(false, a, false, b, (short)0, c, false, false);
}
// C/D (16x16 f32): lane col N=lane&15; VGPR r holds row M = r + 8*(lane>>4).

__device__ __forceinline__ int kxm(int m) { return (m < 16) ? m : (96 + m); } // kept kx rows

__device__ __forceinline__ void cp_lds(uint4* dst, const uint4* src, int n16, int tid) {
  for (int e = tid; e < n16; e += 256) dst[e] = src[e];
}

// ---------------------------------------------------------------------------
// Twiddle setup (f16, ortho 1/sqrt(128) folded in; angle reduced mod 128).
// cEWt [c][w] 32x128 (fwd W-DFT, Bt);  cCT/cST/cSTN [m][h] 32x128 (fwd H, A);
// cCH/cSH/cSHN [h][m] 128x32 (inv H, A);  cIWt [w][r] 128x32 (inv W, A of outT).
// ---------------------------------------------------------------------------
__global__ __launch_bounds__(256) void k_tw(_Float16* cEWt, _Float16* cCT, _Float16* cST,
                                            _Float16* cSTN, _Float16* cCH, _Float16* cSH,
                                            _Float16* cSHN, _Float16* cIWt) {
  const float rs = 0.08838834764831845f;              // 1/sqrt(128)
  const float tp = 6.2831853071795864f / 128.0f;
  for (int e = threadIdx.x; e < 4096; e += 256) {
    { // EWt [c][w]: c<16 -> cos(2pi ky w/128), else -sin
      int c = e >> 7, w = e & 127, ky = c & 15;
      float th = tp * (float)((ky * w) & 127);
      cEWt[e] = (_Float16)(((c < 16) ? cosf(th) : -sinf(th)) * rs);
    }
    { // CT/ST/STN [m][h]
      int m = e >> 7, h = e & 127;
      float ph = tp * (float)((kxm(m) * h) & 127);
      float c = cosf(ph) * rs, s = sinf(ph) * rs;
      cCT[e] = (_Float16)c; cST[e] = (_Float16)s; cSTN[e] = (_Float16)(-s);
    }
    { // CH/SH/SHN [h][m]
      int h = e >> 5, m = e & 31;
      float ph = tp * (float)((kxm(m) * h) & 127);
      float c = cosf(ph) * rs, s = sinf(ph) * rs;
      cCH[e] = (_Float16)c; cSH[e] = (_Float16)s; cSHN[e] = (_Float16)(-s);
    }
    { // IWt [w][r]: r<16 -> a_ky*cos, r>=16 -> -a_ky*sin (a_0=1 else 2;
      // imag of ky=0 bin dropped per C2R semantics -> sin(0)=0 anyway).
      int w = e >> 5, r = e & 31, ky = r & 15;
      float a = (ky == 0) ? 1.0f : 2.0f;
      float th = tp * (float)((ky * w) & 127);
      cIWt[e] = (_Float16)(((r < 16) ? a * cosf(th) : -a * sinf(th)) * rs);
    }
  }
}

// ---------------------------------------------------------------------------
// Weight transpose: w[i][o][m][ky] f32 -> Bt planes [f=m*16+ky][o*32+i] f16,
// plus pre-negated imag plane (f16 WMMA has no A/B negate modifier).
// ---------------------------------------------------------------------------
__global__ __launch_bounds__(256) void k_w(const float* __restrict__ wr,
                                           const float* __restrict__ wi,
                                           _Float16* __restrict__ Wr,
                                           _Float16* __restrict__ Wi,
                                           _Float16* __restrict__ Win) {
  int gid = blockIdx.x * blockDim.x + threadIdx.x;
  if (gid >= NFREQ * PLANE) return;
  int f = gid >> 10, rr = gid & 1023, o = rr >> 5, i = rr & 31, m = f >> 4, ky = f & 15;
  size_t src = (((size_t)(i * 32 + o)) * 32 + m) * 16 + ky;
  Wr[gid] = (_Float16)wr[src];
  float v = wi[src];
  Wi[gid]  = (_Float16)v;
  Win[gid] = (_Float16)(-v);
}

// ---------------------------------------------------------------------------
// Forward: one workgroup per (b,i). Stage A: X1t(32c x 128h) = (x * EW)^store-T;
// Stage B: X2re = CT*X1re + ST*X1im; X2im = CT*X1im + (-ST)*X1re.
// Output planes [f][b*32+i] (2B scatter; merges in 192MB L2).
// ---------------------------------------------------------------------------
__global__ __launch_bounds__(256) void k_fwd(const float* __restrict__ x,
    const _Float16* __restrict__ cEWt, const _Float16* __restrict__ cCT,
    const _Float16* __restrict__ cST, const _Float16* __restrict__ cSTN,
    _Float16* __restrict__ Xr, _Float16* __restrict__ Xi) {
  __shared__ _Float16 sX[128 * 128]  __attribute__((aligned(32)));  // 32 KB
  __shared__ _Float16 sEWt[32 * 128] __attribute__((aligned(32)));  //  8 KB
  __shared__ _Float16 sX1t[32 * 128] __attribute__((aligned(32)));  //  8 KB
  __shared__ _Float16 sCT[32 * 128]  __attribute__((aligned(32)));  //  8 KB
  __shared__ _Float16 sST[32 * 128]  __attribute__((aligned(32)));  //  8 KB
  __shared__ _Float16 sSTN[32 * 128] __attribute__((aligned(32)));  //  8 KB
  const int tid = threadIdx.x, lane = tid & 31, wv = tid >> 5;
  const int wg = blockIdx.x;  // b*32 + i
  const float4* xb4 = (const float4*)(x + (size_t)wg * 16384);
  for (int e = tid; e < 4096; e += 256) {
    float4 v = xb4[e];
    v4h h; h[0] = (_Float16)v.x; h[1] = (_Float16)v.y;
    h[2] = (_Float16)v.z; h[3] = (_Float16)v.w;
    *(v4h*)(sX + 4 * e) = h;
  }
  cp_lds((uint4*)sEWt, (const uint4*)cEWt, 512, tid);
  cp_lds((uint4*)sCT,  (const uint4*)cCT,  512, tid);
  cp_lds((uint4*)sST,  (const uint4*)cST,  512, tid);
  cp_lds((uint4*)sSTN, (const uint4*)cSTN, 512, tid);
  __syncthreads();
  const int g = lane >> 4, n15 = lane & 15;
  // Stage A: 16 tiles (8 h-tiles x 2 c-tiles) over 8 waves; K=128.
#pragma unroll
  for (int t = 0; t < 2; ++t) {
    const int tt = wv * 2 + t, mt = tt >> 1, nt = tt & 1;
    v8f acc = {};
#pragma unroll
    for (int ks = 0; ks < 4; ++ks)
      acc = wmma16(ldsA(sX, mt * 16, ks * 32, 128, lane),
                   ldsBt(sEWt, nt * 16, ks * 32, 128, lane), acc);
    v8h hv;
#pragma unroll
    for (int r = 0; r < 8; ++r) hv[r] = (_Float16)acc[r];
    // transposed store: X1t[col][h], lane's 8 rows are contiguous in h
    *(v8h*)(sX1t + (nt * 16 + n15) * 128 + mt * 16 + 8 * g) = hv;
  }
  __syncthreads();
  // Stage B: X2 (32 kept kx x 16 ky, complex); waves 0-3 (uniform branch).
  if (wv < 4) {
    const int isim = wv >> 1, mt = wv & 1;
    v8f acc = {};
#pragma unroll
    for (int ks = 0; ks < 4; ++ks) {
      if (!isim) {
        acc = wmma16(ldsA(sCT, mt * 16, ks * 32, 128, lane), ldsBt(sX1t, 0, ks * 32, 128, lane), acc);
        acc = wmma16(ldsA(sST, mt * 16, ks * 32, 128, lane), ldsBt(sX1t, 16, ks * 32, 128, lane), acc);
      } else {
        acc = wmma16(ldsA(sCT, mt * 16, ks * 32, 128, lane), ldsBt(sX1t, 16, ks * 32, 128, lane), acc);
        acc = wmma16(ldsA(sSTN, mt * 16, ks * 32, 128, lane), ldsBt(sX1t, 0, ks * 32, 128, lane), acc);
      }
    }
    _Float16* dst = isim ? Xi : Xr;
#pragma unroll
    for (int r = 0; r < 8; ++r) {
      const int m = mt * 16 + r + 8 * g;  // kept-kx index
      dst[(size_t)(m * 16 + n15) * PLANE + wg] = (_Float16)acc[r];
    }
  }
}

// ---------------------------------------------------------------------------
// Mixing: one workgroup per frequency. Or = Xr*Wr + Xi*(-Wi); Oi = Xr*Wi + Xi*Wr.
// A = X [b][i] row-major; B = W planes already transposed [o][i].
// ---------------------------------------------------------------------------
__global__ __launch_bounds__(256) void k_mix(const _Float16* __restrict__ Xr,
    const _Float16* __restrict__ Xi, const _Float16* __restrict__ Wr,
    const _Float16* __restrict__ Wi, const _Float16* __restrict__ Win,
    _Float16* __restrict__ Or_, _Float16* __restrict__ Oi_) {
  __shared__ _Float16 sXr[PLANE] __attribute__((aligned(32)));
  __shared__ _Float16 sXi[PLANE] __attribute__((aligned(32)));
  __shared__ _Float16 sWr[PLANE] __attribute__((aligned(32)));
  __shared__ _Float16 sWi[PLANE] __attribute__((aligned(32)));
  __shared__ _Float16 sWn[PLANE] __attribute__((aligned(32)));
  const int tid = threadIdx.x, lane = tid & 31, wv = tid >> 5;
  const size_t base = (size_t)blockIdx.x * PLANE;
  cp_lds((uint4*)sXr, (const uint4*)(Xr + base), 128, tid);
  cp_lds((uint4*)sXi, (const uint4*)(Xi + base), 128, tid);
  cp_lds((uint4*)sWr, (const uint4*)(Wr + base), 128, tid);
  cp_lds((uint4*)sWi, (const uint4*)(Wi + base), 128, tid);
  cp_lds((uint4*)sWn, (const uint4*)(Win + base), 128, tid);
  __syncthreads();
  const int isim = wv >> 2, mt = (wv >> 1) & 1, nt = wv & 1;
  v8f acc = {};
  if (!isim) {
    acc = wmma16(ldsA(sXr, mt * 16, 0, 32, lane), ldsBt(sWr, nt * 16, 0, 32, lane), acc);
    acc = wmma16(ldsA(sXi, mt * 16, 0, 32, lane), ldsBt(sWn, nt * 16, 0, 32, lane), acc);
  } else {
    acc = wmma16(ldsA(sXr, mt * 16, 0, 32, lane), ldsBt(sWi, nt * 16, 0, 32, lane), acc);
    acc = wmma16(ldsA(sXi, mt * 16, 0, 32, lane), ldsBt(sWr, nt * 16, 0, 32, lane), acc);
  }
  _Float16* dst = isim ? Oi_ : Or_;
  const int g = lane >> 4, n15 = lane & 15;
#pragma unroll
  for (int r = 0; r < 8; ++r)
    dst[base + (mt * 16 + r + 8 * g) * 32 + nt * 16 + n15] = (_Float16)acc[r];
}

// ---------------------------------------------------------------------------
// Inverse: one workgroup per (b,o). Stage D: Gre = CH*Fr + (-SH)*Fi,
// Gim = SH*Fr + CH*Fi -> Z[h][col] (cols 0-15 re, 16-31 im). Stage E computes
// out^T tiles: A = IWt[w][col], Bt = Z[h][col]; lane then owns 8 consecutive w
// -> single v8f (32B) global store per lane.
// ---------------------------------------------------------------------------
__global__ __launch_bounds__(256) void k_inv(const _Float16* __restrict__ Fr_,
    const _Float16* __restrict__ Fi_, const _Float16* __restrict__ cCH,
    const _Float16* __restrict__ cSH, const _Float16* __restrict__ cSHN,
    const _Float16* __restrict__ cIWt, float* __restrict__ out) {
  __shared__ _Float16 sFrT[16 * 32]  __attribute__((aligned(32)));  // [ky][m]
  __shared__ _Float16 sFiT[16 * 32]  __attribute__((aligned(32)));
  __shared__ _Float16 sCH[128 * 32]  __attribute__((aligned(32)));
  __shared__ _Float16 sSH[128 * 32]  __attribute__((aligned(32)));
  __shared__ _Float16 sSHN[128 * 32] __attribute__((aligned(32)));
  __shared__ _Float16 sIWt[128 * 32] __attribute__((aligned(32)));
  __shared__ _Float16 sZ[128 * 32]   __attribute__((aligned(32)));
  const int tid = threadIdx.x, lane = tid & 31, wv = tid >> 5;
  const int wg = blockIdx.x;  // b*32 + o
  for (int e = tid; e < 512; e += 256) {  // f = m*16+ky -> transposed [ky][m]
    const int m = e >> 4, ky = e & 15;
    sFrT[ky * 32 + m] = Fr_[(size_t)e * PLANE + wg];
    sFiT[ky * 32 + m] = Fi_[(size_t)e * PLANE + wg];
  }
  cp_lds((uint4*)sCH,  (const uint4*)cCH,  512, tid);
  cp_lds((uint4*)sSH,  (const uint4*)cSH,  512, tid);
  cp_lds((uint4*)sSHN, (const uint4*)cSHN, 512, tid);
  cp_lds((uint4*)sIWt, (const uint4*)cIWt, 512, tid);
  __syncthreads();
  const int g = lane >> 4, n15 = lane & 15;
  // Stage D: 8 waves = 8 h-tiles, each produces re+im 16x16 tiles.
  {
    const int ht = wv;
    v8f ar = {}, ai = {};
    ar = wmma16(ldsA(sCH,  ht * 16, 0, 32, lane), ldsBt(sFrT, 0, 0, 32, lane), ar);
    ar = wmma16(ldsA(sSHN, ht * 16, 0, 32, lane), ldsBt(sFiT, 0, 0, 32, lane), ar);
    ai = wmma16(ldsA(sSH,  ht * 16, 0, 32, lane), ldsBt(sFrT, 0, 0, 32, lane), ai);
    ai = wmma16(ldsA(sCH,  ht * 16, 0, 32, lane), ldsBt(sFiT, 0, 0, 32, lane), ai);
#pragma unroll
    for (int r = 0; r < 8; ++r) {
      const int h = ht * 16 + r + 8 * g;
      sZ[h * 32 + n15]      = (_Float16)ar[r];
      sZ[h * 32 + 16 + n15] = (_Float16)ai[r];
    }
  }
  __syncthreads();
  // Stage E (out^T): wave wv owns w-tile wv; A fragment reused across 8 h-tiles.
  float* ob = out + (size_t)wg * 16384;
  const v16h a = ldsA(sIWt, wv * 16, 0, 32, lane);
#pragma unroll
  for (int ht = 0; ht < 8; ++ht) {
    v8f acc = {};
    acc = wmma16(a, ldsBt(sZ, ht * 16, 0, 32, lane), acc);
    // lane: h = ht*16+n15 fixed, w = wv*16 + 8g + r contiguous -> 32B store
    *(v8f*)(ob + (ht * 16 + n15) * 128 + wv * 16 + 8 * g) = acc;
  }
}

// ---------------------------------------------------------------------------
extern "C" void kernel_launch(void* const* d_in, const int* in_sizes, int n_in,
                              void* d_out, int out_size, void* d_ws, size_t ws_size,
                              hipStream_t stream) {
  const float* x  = (const float*)d_in[0];
  const float* wr = (const float*)d_in[1];
  const float* wi = (const float*)d_in[2];
  float* out = (float*)d_out;

  char* ws = (char*)d_ws;  // ~7.1 MB total; lives in 192 MB L2
  size_t off = 0;
  auto carve = [&](size_t bytes) {
    char* p = ws + off;
    off += (bytes + 255) & ~(size_t)255;
    return p;
  };
  _Float16* cEWt = (_Float16*)carve(4096 * 2);
  _Float16* cCT  = (_Float16*)carve(4096 * 2);
  _Float16* cST  = (_Float16*)carve(4096 * 2);
  _Float16* cSTN = (_Float16*)carve(4096 * 2);
  _Float16* cCH  = (_Float16*)carve(4096 * 2);
  _Float16* cSH  = (_Float16*)carve(4096 * 2);
  _Float16* cSHN = (_Float16*)carve(4096 * 2);
  _Float16* cIWt = (_Float16*)carve(4096 * 2);
  _Float16* Wr   = (_Float16*)carve((size_t)NFREQ * PLANE * 2);
  _Float16* Wi   = (_Float16*)carve((size_t)NFREQ * PLANE * 2);
  _Float16* Win  = (_Float16*)carve((size_t)NFREQ * PLANE * 2);
  _Float16* Xr   = (_Float16*)carve((size_t)NFREQ * PLANE * 2);
  _Float16* Xi   = (_Float16*)carve((size_t)NFREQ * PLANE * 2);
  _Float16* Or_  = (_Float16*)carve((size_t)NFREQ * PLANE * 2);
  _Float16* Oi_  = (_Float16*)carve((size_t)NFREQ * PLANE * 2);
  (void)in_sizes; (void)n_in; (void)out_size; (void)ws_size;

  hipLaunchKernelGGL(k_tw,  dim3(1),    dim3(256), 0, stream,
                     cEWt, cCT, cST, cSTN, cCH, cSH, cSHN, cIWt);
  hipLaunchKernelGGL(k_w,   dim3(2048), dim3(256), 0, stream, wr, wi, Wr, Wi, Win);
  hipLaunchKernelGGL(k_fwd, dim3(1024), dim3(256), 0, stream,
                     x, cEWt, cCT, cST, cSTN, Xr, Xi);
  hipLaunchKernelGGL(k_mix, dim3(512),  dim3(256), 0, stream,
                     Xr, Xi, Wr, Wi, Win, Or_, Oi_);
  hipLaunchKernelGGL(k_inv, dim3(1024), dim3(256), 0, stream,
                     Or_, Oi_, cCH, cSH, cSHN, cIWt, out);
}
